// head_89249420411453
// MI455X (gfx1250) — matrix operand order repaired and checked by reading, not compiled, not run on hardware
//
#include <hip/hip_runtime.h>

// Problem constants (from reference)
#define TT 256   // block size T
#define CC 384   // embedding size
#define HH 64    // head size

typedef __attribute__((ext_vector_type(16))) __bf16 v16bf;
typedef __attribute__((ext_vector_type(8)))  float  v8f;

// ---- WMMA fragment helpers (A/B use contiguous k = 16*hi + e, not ISA table order; wave32) ----

// A matrix 16x32 bf16, row-major source: lane holds row M = row0 + (lane&15).
// element e: K = 16*hi + e
// -> one contiguous 32-byte run per lane
__device__ __forceinline__ v16bf lds_loadA(const __bf16* base, int ld, int row0,
                                           int col0, int lane) {
  const int hi = (lane >> 4) & 1;
  const __bf16* p = base + (size_t)(row0 + (lane & 15)) * ld + col0;
  v16bf a;
#pragma unroll
  for (int e = 0; e < 16; ++e) a[e] = p[16 * hi + e];
  return a;
}

// B matrix 32x16 where B(k,n) = src[n*ld + k] (src row-major NxK / transposed):
// 16 contiguous bf16 (32 bytes) per lane -> vectorizable ds loads
__device__ __forceinline__ v16bf lds_loadB_nk(const __bf16* base, int ld, int k0,
                                              int n0, int lane) {
  const int hi = (lane >> 4) & 1;
  const __bf16* p = base + (size_t)(n0 + (lane & 15)) * ld + k0 + 16 * hi;
  v16bf b;
#pragma unroll
  for (int e = 0; e < 16; ++e) b[e] = p[e];
  return b;
}

// B fragment straight from fp32 global weights (3 x 96 KB = 288 KB total -> L2-resident)
__device__ __forceinline__ v16bf glb_loadB_kn_f32(const float* base, int ld,
                                                  int k0, int n0, int lane) {
  const int hi = (lane >> 4) & 1;
  const float* p = base + (size_t)(k0 + 16 * hi) * ld + n0 + (lane & 15);
  v16bf b;
#pragma unroll
  for (int e = 0; e < 16; ++e) b[e] = (__bf16)p[(size_t)e * ld];
  return b;
}

__device__ __forceinline__ v8f wmma_bf16(v16bf a, v16bf b, v8f c) {
  return __builtin_amdgcn_wmma_f32_16x16x32_bf16(false, a, false, b, (short)0, c,
                                                 false, false);
}

// pack two fp32 -> packed bf16 dword (lowers to v_cvt_pk_bf16_f32)
__device__ __forceinline__ unsigned pack_bf16x2(float lo, float hi) {
  __bf16 l = (__bf16)lo, h = (__bf16)hi;
  unsigned short ul = __builtin_bit_cast(unsigned short, l);
  unsigned short uh = __builtin_bit_cast(unsigned short, h);
  return (unsigned)ul | ((unsigned)uh << 16);
}

// LDS layout (bytes):
//   [0, 196608)        : X bf16 (256x384)   -- reused later as S (256x256 bf16, t-major)
//   [196608, 229376)   : Q bf16 (256x64, t-major)
//   [229376, 262144)   : K bf16 (256x64, t-major)
//   [262144, 294912)   : Vt bf16 (64x256, h-major = transposed V)
//   [294912, 295936)   : colsum_recip f32[256]
#define LDS_BYTES (196608 + 98304 + 1024)

__global__ __launch_bounds__(256) void attn_head_kernel(
    const float* __restrict__ X, const float* __restrict__ Wk,
    const float* __restrict__ Wq, const float* __restrict__ Wv,
    float* __restrict__ Out) {
  extern __shared__ char smem[];
  __bf16* Xs = (__bf16*)smem;
  __bf16* Ss = (__bf16*)smem;  // overlays Xs (X dead before S written)
  __bf16* Qs = (__bf16*)(smem + 196608);
  __bf16* Ks = (__bf16*)(smem + 196608 + 32768);
  __bf16* Vt = (__bf16*)(smem + 196608 + 65536);   // transposed: Vt[h*256 + t]
  float* csum_r = (float*)(smem + 196608 + 98304); // 1/colsum

  const int tid = threadIdx.x;
  const int lane = tid & 31;
  const int w = tid >> 5;  // wave id, 8 wave32s
  const int hi = (lane >> 4) & 1;
  const int b = blockIdx.x;

  // ---- Phase 1: X fp32 -> bf16 into LDS (float4 global loads, packed stores) ----
  const float4* Xg4 = reinterpret_cast<const float4*>(X + (size_t)b * (TT * CC));
  for (int i = tid; i < (TT * CC) / 4; i += 256) {
    float4 f = Xg4[i];
    uint2 pk;
    pk.x = pack_bf16x2(f.x, f.y);
    pk.y = pack_bf16x2(f.z, f.w);
    *reinterpret_cast<uint2*>(Xs + 4 * i) = pk;
  }
  __syncthreads();

  // ---- Phase 2: K,Q,V = X @ W  (M=256 -> 16 tiles, N=64 -> 4 tiles, K=384) ----
  // wave w owns M-tiles {2w, 2w+1} x all 4 N-tiles; the W B-fragment is loaded
  // ONCE per (nt,kk) and fed to both M accumulators.
  // K,Q stored t-major (phase-3 fragment layouts); V stored h-major
  // (transposed): D-lane's 8 rows are contiguous -> single b128 store, and
  // GEMM2's B fragments become contiguous 32-byte runs.
  const float* Wmat[3] = {Wk, Wq, Wv};
  for (int mtx = 0; mtx < 3; ++mtx) {
    const float* Wp = Wmat[mtx];
    const int mt0 = 2 * w, mt1 = 2 * w + 1;
    for (int nt = 0; nt < 4; ++nt) {
      v8f acc0 = {}, acc1 = {};
#pragma unroll 4
      for (int kk = 0; kk < CC / 32; ++kk) {
        v16bf bb = glb_loadB_kn_f32(Wp, HH, 32 * kk, 16 * nt, lane);  // shared
        v16bf a0 = lds_loadA(Xs, CC, 16 * mt0, 32 * kk, lane);
        v16bf a1 = lds_loadA(Xs, CC, 16 * mt1, 32 * kk, lane);
        acc0 = wmma_bf16(a0, bb, acc0);
        acc1 = wmma_bf16(a1, bb, acc1);
      }
      const int n = 16 * nt + (lane & 15);  // output column (h)
      if (mtx == 2) {
        // V: transposed, packed 16-byte stores (rows contiguous in Vt)
        uint4 pk0, pk1;
        pk0.x = pack_bf16x2(acc0[0], acc0[1]);
        pk0.y = pack_bf16x2(acc0[2], acc0[3]);
        pk0.z = pack_bf16x2(acc0[4], acc0[5]);
        pk0.w = pack_bf16x2(acc0[6], acc0[7]);
        pk1.x = pack_bf16x2(acc1[0], acc1[1]);
        pk1.y = pack_bf16x2(acc1[2], acc1[3]);
        pk1.z = pack_bf16x2(acc1[4], acc1[5]);
        pk1.w = pack_bf16x2(acc1[6], acc1[7]);
        *reinterpret_cast<uint4*>(Vt + (size_t)n * TT + 16 * mt0 + 8 * hi) = pk0;
        *reinterpret_cast<uint4*>(Vt + (size_t)n * TT + 16 * mt1 + 8 * hi) = pk1;
      } else {
        __bf16* Dp = (mtx == 0) ? Ks : Qs;
#pragma unroll
        for (int r = 0; r < 8; ++r) {
          Dp[(size_t)(16 * mt0 + 8 * hi + r) * HH + n] = (__bf16)acc0[r];
          Dp[(size_t)(16 * mt1 + 8 * hi + r) * HH + n] = (__bf16)acc1[r];
        }
      }
    }
  }
  __syncthreads();

  // ---- Phase 3: scores = Q @ K^T * scale, causal mask, exp, column sums ----
  // wave w owns key-tile columns {2w, 2w+1} x all 16 query tiles, so the
  // axis=1 (query-axis) softmax reduction never leaves the wave. Q A-fragments
  // are loaded once per mt and reused for both key columns.
  const float scale = 0.05103103630798287f;  // 384^-0.5 (C, not head size)
  float colpart0 = 0.f, colpart1 = 0.f;
  const int st0 = 2 * w, st1 = 2 * w + 1;
  const int s0c = 16 * st0 + (lane & 15);
  const int s1c = 16 * st1 + (lane & 15);
  for (int mt = 0; mt < 16; ++mt) {
    v16bf a0 = lds_loadA(Qs, HH, 16 * mt, 0, lane);   // shared across st0/st1
    v16bf a1 = lds_loadA(Qs, HH, 16 * mt, 32, lane);
    v8f acc0 = {}, acc1 = {};
    {
      v16bf bb = lds_loadB_nk(Ks, HH, 0, 16 * st0, lane);
      acc0 = wmma_bf16(a0, bb, acc0);
      bb = lds_loadB_nk(Ks, HH, 32, 16 * st0, lane);
      acc0 = wmma_bf16(a1, bb, acc0);
    }
    {
      v16bf bb = lds_loadB_nk(Ks, HH, 0, 16 * st1, lane);
      acc1 = wmma_bf16(a0, bb, acc1);
      bb = lds_loadB_nk(Ks, HH, 32, 16 * st1, lane);
      acc1 = wmma_bf16(a1, bb, acc1);
    }
#pragma unroll
    for (int r = 0; r < 8; ++r) {
      const int t = 16 * mt + 8 * hi + r;
      // tril mask: keep t >= s; exp(-1e9) == 0 exactly
      float e0 = (t >= s0c) ? __expf(acc0[r] * scale) : 0.f;
      float e1 = (t >= s1c) ? __expf(acc1[r] * scale) : 0.f;
      colpart0 += e0;
      colpart1 += e1;
      Ss[(size_t)t * TT + s0c] = (__bf16)e0;  // t-major: phase-5 A layout
      Ss[(size_t)t * TT + s1c] = (__bf16)e1;
    }
  }
  colpart0 += __shfl_xor(colpart0, 16, 32);  // lane L and L^16 share column s
  colpart1 += __shfl_xor(colpart1, 16, 32);
  if (lane < 16) {
    csum_r[s0c] = 1.0f / colpart0;  // store reciprocal
    csum_r[s1c] = 1.0f / colpart1;
  }
  __syncthreads();

  // ---- Phase 4: fold softmax denominator into V: Vt[h][s] *= 1/colsum[s] ----
  // contiguous h-major sweep, multiplies only
  {
    __bf16* vp = Vt + (size_t)tid * 64;  // 16384 elems / 256 threads
    const int sb = (tid & 3) * 64;       // h = tid>>2 fixed, s = sb..sb+63
#pragma unroll 8
    for (int i = 0; i < 64; ++i)
      vp[i] = (__bf16)((float)vp[i] * csum_r[sb + i]);
  }
  __syncthreads();

  // ---- Phase 5: Out = S @ V'  (M=256, N=64, K=256); B shared across M pair ----
  float* Og = Out + (size_t)b * (TT * HH);
  {
    const int mt0 = 2 * w, mt1 = 2 * w + 1;
    for (int nt = 0; nt < 4; ++nt) {
      v8f acc0 = {}, acc1 = {};
#pragma unroll 4
      for (int kk = 0; kk < TT / 32; ++kk) {
        v16bf bb = lds_loadB_nk(Vt, TT, 32 * kk, 16 * nt, lane);  // shared
        v16bf a0 = lds_loadA(Ss, TT, 16 * mt0, 32 * kk, lane);
        v16bf a1 = lds_loadA(Ss, TT, 16 * mt1, 32 * kk, lane);
        acc0 = wmma_bf16(a0, bb, acc0);
        acc1 = wmma_bf16(a1, bb, acc1);
      }
      const int n = 16 * nt + (lane & 15);
#pragma unroll
      for (int r = 0; r < 8; ++r) {
        Og[(size_t)(16 * mt0 + 8 * hi + r) * HH + n] = acc0[r];
        Og[(size_t)(16 * mt1 + 8 * hi + r) * HH + n] = acc1[r];
      }
    }
  }
}

extern "C" void kernel_launch(void* const* d_in, const int* in_sizes, int n_in,
                              void* d_out, int out_size, void* d_ws,
                              size_t ws_size, hipStream_t stream) {
  const float* X = (const float*)d_in[0];
  const float* Wk = (const float*)d_in[1];
  const float* Wq = (const float*)d_in[2];
  const float* Wv = (const float*)d_in[3];
  float* Out = (float*)d_out;

  const int B = in_sizes[0] / (TT * CC);  // 512

  // Opt into >64KB dynamic LDS (WGP has 320KB on CDNA5). Host-side attribute
  // set is deterministic and capture-safe (not a stream operation).
  hipFuncSetAttribute((const void*)attn_head_kernel,
                      hipFuncAttributeMaxDynamicSharedMemorySize, LDS_BYTES);

  attn_head_kernel<<<B, 256, LDS_BYTES, stream>>>(X, Wk, Wq, Wv, Out);
}
